// AttentionBlock_10024453669669
// MI455X (gfx1250) — matrix operand order repaired and checked by reading, not compiled, not run on hardware
//
#include <hip/hip_runtime.h>
#include <hip/hip_bf16.h>

// ---------------- problem constants ----------------
#define BB   4
#define NSEQ 2048
#define CDIM 1024
#define HEADS 16
#define DHEAD 64
#define EMBD 1024
#define MROWS (BB * NSEQ)      // 8192
#define EPS 1e-5f

// ---------------- WMMA fragment types (CDNA5, wave32) ----------------
typedef __attribute__((ext_vector_type(16))) __bf16 v16bf;
typedef __attribute__((ext_vector_type(8)))  float  v8f;

union Frag {
    v16bf bf;
    uint4 q[2];
};

__device__ __forceinline__ v8f wmma_bf16(const Frag& a, const Frag& b, v8f c) {
    // D = A(16x32 bf16) * B(32x16 bf16) + C(16x16 f32)
    return __builtin_amdgcn_wmma_f32_16x16x32_bf16(
        /*neg_a=*/false, a.bf, /*neg_b=*/false, b.bf,
        /*c_mod=*/(short)0, c, /*reuse_a=*/false, /*reuse_b=*/false);
}

__device__ __forceinline__ unsigned short f2bf(float f) {
    unsigned int u = __float_as_uint(f);
    unsigned int r = u + 0x7FFFu + ((u >> 16) & 1u);   // round-to-nearest-even
    return (unsigned short)(r >> 16);
}

// ---------------- elementwise fp32 -> bf16 ----------------
__global__ __launch_bounds__(256) void cvt_bf16(const float* __restrict__ in,
                                                unsigned short* __restrict__ out) {
    int i = blockIdx.x * 256 + threadIdx.x;
    out[i] = f2bf(in[i]);
}

// ---------------- LayerNorm row stats over C=1024 ----------------
__global__ __launch_bounds__(256) void rowstats(const float* __restrict__ x,
                                                float* __restrict__ mu,
                                                float* __restrict__ rvar) {
    int r = blockIdx.x;
    const float* xr = x + (size_t)r * CDIM;
    float s = 0.f, ss = 0.f;
    #pragma unroll
    for (int i = 0; i < 4; ++i) {
        float v = xr[threadIdx.x + i * 256];
        s += v; ss += v * v;
    }
    #pragma unroll
    for (int off = 16; off; off >>= 1) {
        s  += __shfl_xor(s,  off, 32);
        ss += __shfl_xor(ss, off, 32);
    }
    __shared__ float sb[8], ssb[8];
    int wave = threadIdx.x >> 5, lane = threadIdx.x & 31;
    if (lane == 0) { sb[wave] = s; ssb[wave] = ss; }
    __syncthreads();
    if (threadIdx.x == 0) {
        float S = 0.f, SS = 0.f;
        #pragma unroll
        for (int w = 0; w < 8; ++w) { S += sb[w]; SS += ssb[w]; }
        float m = S * (1.f / CDIM);
        float var = SS * (1.f / CDIM) - m * m;
        mu[r] = m;
        rvar[r] = rsqrtf(var + EPS);
    }
}

// ---------------- FiLM: h = ln(x)*(1+scale)+shift -> bf16 ----------------
__global__ __launch_bounds__(256) void film(const float* __restrict__ x,
                                            const float* __restrict__ mu,
                                            const float* __restrict__ rvar,
                                            const float* __restrict__ se,
                                            unsigned short* __restrict__ hB) {
    int r = blockIdx.x;
    float m = mu[r], rv = rvar[r];
    const float* xr  = x  + (size_t)r * CDIM;
    const float* ser = se + (size_t)r * (2 * CDIM);
    unsigned short* hr = hB + (size_t)r * CDIM;
    #pragma unroll
    for (int i = 0; i < 4; ++i) {
        int c = threadIdx.x + i * 256;
        float v = (xr[c] - m) * rv * (1.f + ser[c]) + ser[CDIM + c];
        hr[c] = f2bf(v);
    }
}

// ---------------- per-head q/k LN + layout, V transpose ----------------
// qkv row layout: [3][H][D]; outputs q (pre-scaled by 1/sqrt(D)), k as (B*H, N, D) bf16,
// vt as (B*H, D, N) bf16
__global__ __launch_bounds__(256) void qkln(const float* __restrict__ qkv,
                                            unsigned short* __restrict__ qB,
                                            unsigned short* __restrict__ kB,
                                            unsigned short* __restrict__ vtB) {
    int r = blockIdx.x;              // r = b*N + n
    int b = r >> 11, n = r & (NSEQ - 1);
    int wave = threadIdx.x >> 5, lane = threadIdx.x & 31;
    const float* row = qkv + (size_t)r * (3 * CDIM);
    #pragma unroll
    for (int hh = 0; hh < 2; ++hh) {
        int h = wave * 2 + hh;
        #pragma unroll
        for (int part = 0; part < 2; ++part) {
            const float* p = row + part * CDIM + h * DHEAD;
            float x0 = p[lane], x1 = p[lane + 32];
            float s = x0 + x1, ss = x0 * x0 + x1 * x1;
            #pragma unroll
            for (int off = 16; off; off >>= 1) {
                s  += __shfl_xor(s,  off, 32);
                ss += __shfl_xor(ss, off, 32);
            }
            float m = s * (1.f / DHEAD);
            float var = ss * (1.f / DHEAD) - m * m;
            float rv = rsqrtf(var + EPS);
            if (part == 0) rv *= 0.125f;    // fold 1/sqrt(64) attention scale into q
            unsigned short* dst = (part == 0 ? qB : kB) +
                ((size_t)(b * HEADS + h) * NSEQ + n) * DHEAD;
            dst[lane]      = f2bf((x0 - m) * rv);
            dst[lane + 32] = f2bf((x1 - m) * rv);
        }
    }
    #pragma unroll
    for (int i = 0; i < 4; ++i) {
        int c = threadIdx.x + i * 256;
        int h = c >> 6, d = c & 63;
        vtB[((size_t)(b * HEADS + h) * DHEAD + d) * NSEQ + n] = f2bf(row[2 * CDIM + c]);
    }
}

// ---------------- generic bf16 GEMM: C = A(MxK) * W(NcxK)^T (+bias)(+res) ----------------
// block: 256 thr = 8 waves arranged 4(M) x 2(N); wave tile 32x64; block tile 128x128
__global__ __launch_bounds__(256) void gemm_bf16(const unsigned short* __restrict__ A,
                                                 const unsigned short* __restrict__ W,
                                                 const float* __restrict__ bias,
                                                 const float* __restrict__ residual,
                                                 float* __restrict__ C,
                                                 int Nc, int K) {
    int wave = threadIdx.x >> 5, lane = threadIdx.x & 31;
    int lr = lane & 15, lh = lane >> 4;
    int m0 = blockIdx.y * 128 + (wave >> 1) * 32;
    int n0 = blockIdx.x * 128 + (wave & 1) * 64;

    const unsigned short* ap[2];
    #pragma unroll
    for (int mi = 0; mi < 2; ++mi)
        ap[mi] = A + (size_t)(m0 + mi * 16 + lr) * K + lh * 16;
    const unsigned short* bp[4];
    #pragma unroll
    for (int ni = 0; ni < 4; ++ni)
        bp[ni] = W + (size_t)(n0 + ni * 16 + lr) * K + lh * 16;

    v8f acc[2][4] = {};
    for (int k = 0; k < K; k += 32) {
        Frag a[2], bf[4];
        #pragma unroll
        for (int mi = 0; mi < 2; ++mi) {
            a[mi].q[0] = *(const uint4*)(ap[mi] + k);
            a[mi].q[1] = *(const uint4*)(ap[mi] + k + 8);
        }
        #pragma unroll
        for (int ni = 0; ni < 4; ++ni) {
            bf[ni].q[0] = *(const uint4*)(bp[ni] + k);
            bf[ni].q[1] = *(const uint4*)(bp[ni] + k + 8);
        }
        #pragma unroll
        for (int mi = 0; mi < 2; ++mi)
            #pragma unroll
            for (int ni = 0; ni < 4; ++ni)
                acc[mi][ni] = wmma_bf16(a[mi], bf[ni], acc[mi][ni]);
    }
    #pragma unroll
    for (int mi = 0; mi < 2; ++mi)
        #pragma unroll
        for (int ni = 0; ni < 4; ++ni) {
            int n = n0 + ni * 16 + lr;
            float bv = bias ? bias[n] : 0.f;
            #pragma unroll
            for (int v = 0; v < 8; ++v) {
                int m = m0 + mi * 16 + lh * 8 + v;
                size_t idx = (size_t)m * Nc + n;
                float val = acc[mi][ni][v] + bv;
                if (residual) val += residual[idx];
                C[idx] = val;
            }
        }
}

// ---------------- flash attention: per wave 16 queries, keys streamed x32 ----------------
__global__ __launch_bounds__(256) void attn(const unsigned short* __restrict__ qB,
                                            const unsigned short* __restrict__ kB,
                                            const unsigned short* __restrict__ vtB,
                                            unsigned short* __restrict__ oB,
                                            float* __restrict__ oF) {
    int bh = blockIdx.y;                 // b*H + h
    int b = bh >> 4, h = bh & 15;
    int wave = threadIdx.x >> 5, lane = threadIdx.x & 31;
    int lr = lane & 15, lh = lane >> 4;
    int q0 = blockIdx.x * 128 + wave * 16;

    const unsigned short* qp = qB  + ((size_t)bh * NSEQ + q0) * DHEAD;
    const unsigned short* kp = kB  + (size_t)bh * NSEQ * DHEAD;
    const unsigned short* vp = vtB + (size_t)bh * DHEAD * NSEQ;

    __shared__ __align__(16) unsigned short ps[8][16 * 32];   // 8 KB, 1 KB/wave
    unsigned short* myps = ps[wave];

    // Q fragments: 16x32 for d=[0,32) and d=[32,64)  (q is pre-scaled by 1/sqrt(D))
    Frag qa0, qa1;
    {
        const unsigned short* qr = qp + lr * DHEAD + lh * 16;
        qa0.q[0] = *(const uint4*)(qr);      qa0.q[1] = *(const uint4*)(qr + 8);
        qa1.q[0] = *(const uint4*)(qr + 32); qa1.q[1] = *(const uint4*)(qr + 40);
    }

    v8f oacc[4] = {};
    float mrun[8], lrun[8];
    #pragma unroll
    for (int v = 0; v < 8; ++v) { mrun[v] = -1e30f; lrun[v] = 0.f; }

    for (int kk = 0; kk < NSEQ; kk += 32) {
        v8f s0 = {}, s1 = {};
        {
            const unsigned short* kr0 = kp + (size_t)(kk + lr)      * DHEAD + lh * 16;
            const unsigned short* kr1 = kp + (size_t)(kk + 16 + lr) * DHEAD + lh * 16;
            Frag kb;
            kb.q[0] = *(const uint4*)(kr0);      kb.q[1] = *(const uint4*)(kr0 + 8);
            s0 = wmma_bf16(qa0, kb, s0);
            kb.q[0] = *(const uint4*)(kr0 + 32); kb.q[1] = *(const uint4*)(kr0 + 40);
            s0 = wmma_bf16(qa1, kb, s0);
            kb.q[0] = *(const uint4*)(kr1);      kb.q[1] = *(const uint4*)(kr1 + 8);
            s1 = wmma_bf16(qa0, kb, s1);
            kb.q[0] = *(const uint4*)(kr1 + 32); kb.q[1] = *(const uint4*)(kr1 + 40);
            s1 = wmma_bf16(qa1, kb, s1);
        }
        // online softmax over this 32-key block; row m = lh*8+v lives in a 16-lane half
        float corr[8];
        #pragma unroll
        for (int v = 0; v < 8; ++v) {
            float a  = s0[v], c2 = s1[v];
            float mx = fmaxf(a, c2);
            #pragma unroll
            for (int off = 1; off < 16; off <<= 1) mx = fmaxf(mx, __shfl_xor(mx, off, 32));
            float nm = fmaxf(mrun[v], mx);
            float p0 = __expf(a - nm), p1 = __expf(c2 - nm);
            float rs = p0 + p1;
            #pragma unroll
            for (int off = 1; off < 16; off <<= 1) rs += __shfl_xor(rs, off, 32);
            corr[v] = __expf(mrun[v] - nm);
            lrun[v] = lrun[v] * corr[v] + rs;
            mrun[v] = nm;
            int rowoff = (lh * 8 + v) * 32;           // D-layout -> row-major scratch
            myps[rowoff + lr]      = f2bf(p0);
            myps[rowoff + 16 + lr] = f2bf(p1);
        }
        #pragma unroll
        for (int j = 0; j < 4; ++j)
            #pragma unroll
            for (int v = 0; v < 8; ++v) oacc[j][v] *= corr[v];

        // re-read probs as A-fragment (same-wave LDS, in-order DS)
        Frag pa;
        pa.q[0] = *(const uint4*)&myps[lr * 32 + lh * 16];
        pa.q[1] = *(const uint4*)&myps[lr * 32 + lh * 16 + 8];
        #pragma unroll
        for (int j = 0; j < 4; ++j) {
            const unsigned short* vr = vp + (size_t)(j * 16 + lr) * NSEQ + kk + lh * 16;
            Frag vb;
            vb.q[0] = *(const uint4*)(vr); vb.q[1] = *(const uint4*)(vr + 8);
            oacc[j] = wmma_bf16(pa, vb, oacc[j]);
        }
    }
    // finalize: o = acc / l ; write (B,N,H*D) fp32 + bf16
    #pragma unroll
    for (int v = 0; v < 8; ++v) {
        float inv = 1.f / lrun[v];
        int nq = q0 + lh * 8 + v;
        #pragma unroll
        for (int j = 0; j < 4; ++j) {
            float val = oacc[j][v] * inv;
            size_t idx = ((size_t)b * NSEQ + nq) * CDIM + h * DHEAD + j * 16 + lr;
            oF[idx] = val;
            oB[idx] = f2bf(val);
        }
    }
}

// ---------------- launcher ----------------
extern "C" void kernel_launch(void* const* d_in, const int* in_sizes, int n_in,
                              void* d_out, int out_size, void* d_ws, size_t ws_size,
                              hipStream_t stream) {
    (void)in_sizes; (void)n_in; (void)out_size; (void)ws_size;
    const float* x      = (const float*)d_in[0];
    const float* emb    = (const float*)d_in[1];
    const float* W_emb  = (const float*)d_in[2];
    const float* b_emb  = (const float*)d_in[3];
    const float* W_proj = (const float*)d_in[4];
    const float* W_out  = (const float*)d_in[5];
    float* out = (float*)d_out;

    char* ws = (char*)d_ws;
    size_t off = 0;
    auto alloc = [&](size_t bytes) -> void* {
        void* p = ws + off;
        off += (bytes + 255) & ~(size_t)255;
        return p;
    };
    // Region 0: embB (live through GEMM1), then reused as hB (FiLM -> GEMM2)
    unsigned short* embB   = (unsigned short*)alloc((size_t)MROWS * EMBD * 2);    // 16 MB
    unsigned short* hB     = embB;                                                // alias
    unsigned short* WembB  = (unsigned short*)alloc((size_t)2 * CDIM * EMBD * 2); // 4 MB
    unsigned short* WprojB = (unsigned short*)alloc((size_t)3 * CDIM * CDIM * 2); // 6 MB
    unsigned short* WoutB  = (unsigned short*)alloc((size_t)CDIM * CDIM * 2);     // 2 MB
    float* mu   = (float*)alloc((size_t)MROWS * 4);
    float* rvar = (float*)alloc((size_t)MROWS * 4);
    // Region 1: se (64 MB, dead after film) / qkv (96 MB, dead after qkln) /
    //           oB+oF (48 MB, written by attn after qkv is dead)
    char* region1 = (char*)alloc((size_t)MROWS * 3 * CDIM * 4);                   // 96 MB
    float* se  = (float*)region1;
    float* qkv = (float*)region1;
    unsigned short* oB = (unsigned short*)region1;
    float* oF = (float*)(region1 + (size_t)MROWS * CDIM * 2);
    // Region 2: q/k/vt (live qkln -> attn)
    unsigned short* qBuf  = (unsigned short*)alloc((size_t)MROWS * CDIM * 2);     // 16 MB
    unsigned short* kBuf  = (unsigned short*)alloc((size_t)MROWS * CDIM * 2);     // 16 MB
    unsigned short* vtBuf = (unsigned short*)alloc((size_t)MROWS * CDIM * 2);     // 16 MB

    // 1) downconvert activations + weights to bf16
    cvt_bf16<<<(MROWS * EMBD) / 256, 256, 0, stream>>>(emb, embB);
    cvt_bf16<<<(2 * CDIM * EMBD) / 256, 256, 0, stream>>>(W_emb, WembB);
    cvt_bf16<<<(3 * CDIM * CDIM) / 256, 256, 0, stream>>>(W_proj, WprojB);
    cvt_bf16<<<(CDIM * CDIM) / 256, 256, 0, stream>>>(W_out, WoutB);
    // 2) LN stats for x
    rowstats<<<MROWS, 256, 0, stream>>>(x, mu, rvar);
    // 3) se = emb @ W_emb^T + b_emb  (wait: film reads se; hB aliases embB which is
    //    an input here, so GEMM1 must fully precede film -- stream order guarantees it)
    gemm_bf16<<<dim3((2 * CDIM) / 128, MROWS / 128), 256, 0, stream>>>(
        embB, WembB, b_emb, nullptr, se, 2 * CDIM, EMBD);
    // 4) FiLM -> h (bf16, overwrites embB region)
    film<<<MROWS, 256, 0, stream>>>(x, mu, rvar, se, hB);
    // 5) qkv = h @ W_proj^T  (overwrites se region; se dead)
    gemm_bf16<<<dim3((3 * CDIM) / 128, MROWS / 128), 256, 0, stream>>>(
        hB, WprojB, nullptr, nullptr, qkv, 3 * CDIM, CDIM);
    // 6) per-head q/k LN + head layout + V transpose
    qkln<<<MROWS, 256, 0, stream>>>(qkv, qBuf, kBuf, vtBuf);
    // 7) attention (writes oB/oF over qkv region; qkv dead)
    attn<<<dim3(NSEQ / 128, BB * HEADS), 256, 0, stream>>>(qBuf, kBuf, vtBuf, oB, oF);
    // 8) out = o + o @ W_out^T
    gemm_bf16<<<dim3(CDIM / 128, MROWS / 128), 256, 0, stream>>>(
        oB, WoutB, nullptr, oF, out, CDIM, CDIM);
}
